// MetaLinear3_22565758174013
// MI455X (gfx1250) — matrix operand order repaired and checked by reading, not compiled, not run on hardware
//
#include <hip/hip_runtime.h>
#include <hip/hip_bf16.h>

// MI455X / gfx1250, wave32. Three bf16x2-split WMMA GEMMs + analytic epilogue.

typedef __bf16 bf16;
typedef __attribute__((ext_vector_type(16))) __bf16 v16bf;
typedef __attribute__((ext_vector_type(8)))  __bf16 v8bf;
typedef __attribute__((ext_vector_type(8)))  float   v8f;

union V16 { v16bf v; v8bf h[2]; };

#define TOKENS 1024   // B*S = 2*512
#define DDIM   512    // D_IN == D_OUT
#define EPSLN  1e-5f

// ---------------------------------------------------------------------------
// Kernel 0: split fp32 -> (bf16 hi, bf16 lo) with RNE; x ~= hi + lo to ~2^-18.
// ---------------------------------------------------------------------------
__global__ void split_bf16_kernel(const float* __restrict__ src,
                                  bf16* __restrict__ hi,
                                  bf16* __restrict__ lo, int n) {
  int i = blockIdx.x * blockDim.x + threadIdx.x;
  int stride = gridDim.x * blockDim.x;
  for (; i < n; i += stride) {
    float v = src[i];
    bf16 h = (bf16)v;
    float r = v - (float)h;
    hi[i] = h;
    lo[i] = (bf16)r;
  }
}

// ---------------------------------------------------------------------------
// Kernel 1: fused triple GEMM  O{1,2,b}[t,e] = sum_d x[t,d] * W{in,out,b}[e,d]
// One wave per 16x16 output tile (shared across the 3 outputs), K=512 in
// steps of 32 via V_WMMA_F32_16X16X32_BF16, 3-term bf16x2 split per product.
// grid = (64 M-tiles, 4), block = 256 threads = 8 waves (ntile = by*8+wave).
// ---------------------------------------------------------------------------
__device__ __forceinline__ v8f wmma_bf16(const v16bf& a, const v16bf& b, v8f c) {
  return __builtin_amdgcn_wmma_f32_16x16x32_bf16(
      /*neg_a=*/false, a, /*neg_b=*/false, b,
      /*c_mod=*/(short)0, c, /*reuse_a=*/false, /*reuse_b=*/false);
}

__global__ __launch_bounds__(256)
void gemm3_wmma_kernel(const bf16* __restrict__ xh, const bf16* __restrict__ xl,
                       const bf16* __restrict__ aWh, const bf16* __restrict__ aWl,
                       const bf16* __restrict__ bWh, const bf16* __restrict__ bWl,
                       const bf16* __restrict__ cWh, const bf16* __restrict__ cWl,
                       float* __restrict__ o1, float* __restrict__ o2,
                       float* __restrict__ ob) {
  const int lane  = threadIdx.x & 31;
  const int wave  = threadIdx.x >> 5;
  const int mtile = blockIdx.x;
  const int ntile = blockIdx.y * 8 + wave;
  const int m  = lane & 15;     // A: row within tile / B: column within tile
  const int hh = lane >> 4;     // half-wave selector

  // A rows: x[token, :] ; lane m+16h holds row m, K 16h..16h+15
  const bf16* aHp = xh + (size_t)(mtile * 16 + m) * DDIM;
  const bf16* aLp = xl + (size_t)(mtile * 16 + m) * DDIM;
  // B columns: Bt[k,n] = W[ntile*16+n, k] -> contiguous W row per lane.
  const size_t crow = (size_t)(ntile * 16 + m) * DDIM;
  const bf16* b1H = aWh + crow; const bf16* b1L = aWl + crow;
  const bf16* b2H = bWh + crow; const bf16* b2L = bWl + crow;
  const bf16* b3H = cWh + crow; const bf16* b3L = cWl + crow;

  v8f acc1 = {}; v8f acc2 = {}; v8f accb = {};

  for (int k0 = 0; k0 < DDIM; k0 += 32) {
    V16 ah, al;
    ah.h[0] = *(const v8bf*)(aHp + k0 + 16 * hh);
    ah.h[1] = *(const v8bf*)(aHp + k0 + 16 * hh + 8);
    al.h[0] = *(const v8bf*)(aLp + k0 + 16 * hh);
    al.h[1] = *(const v8bf*)(aLp + k0 + 16 * hh + 8);

    const int bo = k0 + 16 * hh;   // lanes 0-15: K 0..15, lanes 16-31: K 16..31

    V16 bh, bl;
    bh.h[0] = *(const v8bf*)(b1H + bo); bh.h[1] = *(const v8bf*)(b1H + bo + 8);
    bl.h[0] = *(const v8bf*)(b1L + bo); bl.h[1] = *(const v8bf*)(b1L + bo + 8);
    acc1 = wmma_bf16(ah.v, bh.v, acc1);
    acc1 = wmma_bf16(ah.v, bl.v, acc1);
    acc1 = wmma_bf16(al.v, bh.v, acc1);

    bh.h[0] = *(const v8bf*)(b2H + bo); bh.h[1] = *(const v8bf*)(b2H + bo + 8);
    bl.h[0] = *(const v8bf*)(b2L + bo); bl.h[1] = *(const v8bf*)(b2L + bo + 8);
    acc2 = wmma_bf16(ah.v, bh.v, acc2);
    acc2 = wmma_bf16(ah.v, bl.v, acc2);
    acc2 = wmma_bf16(al.v, bh.v, acc2);

    bh.h[0] = *(const v8bf*)(b3H + bo); bh.h[1] = *(const v8bf*)(b3H + bo + 8);
    bl.h[0] = *(const v8bf*)(b3L + bo); bl.h[1] = *(const v8bf*)(b3L + bo + 8);
    accb = wmma_bf16(ah.v, bh.v, accb);
    accb = wmma_bf16(ah.v, bl.v, accb);
    accb = wmma_bf16(al.v, bh.v, accb);
  }

  // D layout: VGPR g, lanes 0-15 -> row g, lanes 16-31 -> row g+8; col = lane%16
  const int col  = ntile * 16 + m;
  const int row0 = mtile * 16 + hh * 8;
#pragma unroll
  for (int g = 0; g < 8; ++g) {
    const size_t idx = (size_t)(row0 + g) * DDIM + col;
    o1[idx] = acc1[g];
    o2[idx] = acc2[g];
    ob[idx] = accb[g];
  }
}

// ---------------------------------------------------------------------------
// Kernel 2: per-token epilogue. One wave32 per token, shfl_xor reductions.
// y[i] = w2[i]*s*rsqrt(w2[i]^2*var1+eps) + (bm[i]-mu_b)*rsqrt(var_b+eps)
// with s = dot(w1,x) - mean(w1)*sum(x).
// ---------------------------------------------------------------------------
__global__ __launch_bounds__(256)
void combine_kernel(const float* __restrict__ W1, const float* __restrict__ W2,
                    const float* __restrict__ Bm, const float* __restrict__ x,
                    float* __restrict__ y) {
  const int lane = threadIdx.x & 31;
  const int t    = blockIdx.x * 8 + (threadIdx.x >> 5);
  const float* w1 = W1 + (size_t)t * DDIM;
  const float* w2 = W2 + (size_t)t * DDIM;
  const float* bm = Bm + (size_t)t * DDIM;
  const float* xr = x  + (size_t)t * DDIM;

  float s1 = 0.f, q1 = 0.f, dx = 0.f, sx = 0.f, sb = 0.f, qb = 0.f;
  for (int j = lane; j < DDIM; j += 32) {
    float a = w1[j], xx = xr[j], bb = bm[j];
    s1 += a; q1 += a * a; dx += a * xx; sx += xx; sb += bb; qb += bb * bb;
  }
#pragma unroll
  for (int off = 16; off > 0; off >>= 1) {
    s1 += __shfl_xor(s1, off, 32);
    q1 += __shfl_xor(q1, off, 32);
    dx += __shfl_xor(dx, off, 32);
    sx += __shfl_xor(sx, off, 32);
    sb += __shfl_xor(sb, off, 32);
    qb += __shfl_xor(qb, off, 32);
  }
  const float inv = 1.0f / (float)DDIM;
  const float mu1  = s1 * inv;
  const float var1 = q1 * inv - mu1 * mu1;
  const float s    = dx - mu1 * sx;
  const float mub  = sb * inv;
  const float varb = qb * inv - mub * mub;
  const float rb   = rsqrtf(varb + EPSLN);

  float* yo = y + (size_t)t * DDIM;
  for (int i = lane; i < DDIM; i += 32) {
    float v2 = w2[i];
    yo[i] = v2 * s * rsqrtf(v2 * v2 * var1 + EPSLN) + (bm[i] - mub) * rb;
  }
}

// ---------------------------------------------------------------------------
extern "C" void kernel_launch(void* const* d_in, const int* in_sizes, int n_in,
                              void* d_out, int out_size, void* d_ws, size_t ws_size,
                              hipStream_t stream) {
  (void)in_sizes; (void)n_in; (void)out_size; (void)ws_size;
  const float* x    = (const float*)d_in[0];   // [1024, 512]
  const float* Win  = (const float*)d_in[1];   // [512, 512]
  const float* Wout = (const float*)d_in[2];   // [512, 512]
  const float* Wb   = (const float*)d_in[3];   // [512, 512]
  float* y = (float*)d_out;                    // [1024, 512]

  char* ws = (char*)d_ws;
  size_t off = 0;
  const size_t NX = (size_t)TOKENS * DDIM;   // 524288
  const size_t NW = (size_t)DDIM * DDIM;     // 262144

  bf16* xh  = (bf16*)(ws + off); off += NX * sizeof(bf16);
  bf16* xl  = (bf16*)(ws + off); off += NX * sizeof(bf16);
  bf16* wih = (bf16*)(ws + off); off += NW * sizeof(bf16);
  bf16* wil = (bf16*)(ws + off); off += NW * sizeof(bf16);
  bf16* woh = (bf16*)(ws + off); off += NW * sizeof(bf16);
  bf16* wol = (bf16*)(ws + off); off += NW * sizeof(bf16);
  bf16* wbh = (bf16*)(ws + off); off += NW * sizeof(bf16);
  bf16* wbl = (bf16*)(ws + off); off += NW * sizeof(bf16);
  float* O1 = (float*)(ws + off); off += NX * sizeof(float);
  float* O2 = (float*)(ws + off); off += NX * sizeof(float);
  float* OB = (float*)(ws + off); off += NX * sizeof(float);

  split_bf16_kernel<<<512, 256, 0, stream>>>(x,    xh,  xl,  (int)NX);
  split_bf16_kernel<<<256, 256, 0, stream>>>(Win,  wih, wil, (int)NW);
  split_bf16_kernel<<<256, 256, 0, stream>>>(Wout, woh, wol, (int)NW);
  split_bf16_kernel<<<256, 256, 0, stream>>>(Wb,   wbh, wbl, (int)NW);

  // 64 M-tiles x (4*8)=32 N-tiles; 8 waves per block.
  gemm3_wmma_kernel<<<dim3(64, 4), 256, 0, stream>>>(
      xh, xl, wih, wil, woh, wol, wbh, wbl, O1, O2, OB);

  combine_kernel<<<TOKENS / 8, 256, 0, stream>>>(O1, O2, OB, x, y);
}
